// ClampedHungarianLoss_64003602645342
// MI455X (gfx1250) — compile-verified, matrix operand order checked by reading, not run on hardware
//
#include <hip/hip_runtime.h>
#include <hip/hip_bf16.h>

// ---------------------------------------------------------------------------
// ClampedHungarianLoss for MI455X (gfx1250)
//
// Strategy (latency-bound problem):
//  * kernel 1: 16 blocks x 32 threads (one wave32 per batch, one WGP each).
//      - TDM (tensor_load_to_lds) stages the full 256x256 f32 cost matrix
//        into LDS (256KB of the WGP's 320KB), then clamp-to-20 in place.
//      - Jonker-Volgenant shortest augmenting path, wave-synchronous:
//        each lane owns 8 columns (minv/v/used in VGPRs), argmin via
//        __shfl_xor tree; u/p/way scatter-indexed arrays in LDS.
//      - Writes the 256 matched (UNclamped) costs per batch to d_ws.
//  * kernel 2: one wave, v_wmma_f32_16x16x32_f16 reduction of the 4096
//      matched values (B = ones => every C column equals the total sum).
// ---------------------------------------------------------------------------

#define NN     256
#define BATCH  16
#define MAXC   20.0f
#define CST_ELEMS (NN * NN)

typedef __attribute__((ext_vector_type(16))) _Float16     v16h;
typedef __attribute__((ext_vector_type(8)))  float        v8f;
typedef __attribute__((ext_vector_type(4)))  unsigned int u32x4;
typedef __attribute__((ext_vector_type(8)))  int          i32x8;
typedef __attribute__((ext_vector_type(4)))  int          i32x4;

#if defined(__has_builtin)
#if __has_builtin(__builtin_amdgcn_tensor_load_to_lds) && \
    __has_builtin(__builtin_amdgcn_s_wait_tensorcnt)
#define HAVE_TDM 1
#endif
#endif

// --- compile-only probes: these show up in stderr even on success ---
#if defined(HAVE_TDM)
#warning "PROBE: TDM builtin path ENABLED (tensor_load_to_lds + s_wait_tensorcnt)"
#else
#warning "PROBE: TDM builtin NOT available -- using plain global->LDS fallback"
#endif
#if __clang_major__ >= 23
#warning "PROBE: clang >= 23 (6-arg tensor_load_to_lds form)"
#else
#warning "PROBE: clang < 23 (5-arg tensor_load_to_lds form)"
#endif

// LDS layout: cst[65536] f32 | u[257] f32 | p[257] i32 | way[257] i32 | colofrow[256] i32
#define SMEM_BYTES ((CST_ELEMS + (NN + 1)) * 4 + 2 * (NN + 1) * 4 + NN * 4)

__global__ __launch_bounds__(32)
void hungarian_kernel(const float* __restrict__ D, float* __restrict__ matched_ws) {
    extern __shared__ char smem[];
    float* cst      = (float*)smem;
    float* u        = cst + CST_ELEMS;
    int*   p        = (int*)(u + (NN + 1));
    int*   way      = p + (NN + 1);
    int*   colofrow = way + (NN + 1);

    const int   b    = blockIdx.x;
    const int   lane = threadIdx.x;                  // wave32: 0..31
    const float* Db  = D + (size_t)b * CST_ELEMS;

    // ---- Stage cost matrix into LDS (TDM if available), clamp to MAX_COST ----
#if defined(HAVE_TDM)
    {
        unsigned long long ga = (unsigned long long)(const void*)Db;
        unsigned ldsaddr = (unsigned)(unsigned long long)(void*)cst; // low 32 bits of LDS aperture = LDS byte addr
        u32x4 g0;
        g0[0] = 1u;                                   // count = 1 (valid descriptor)
        g0[1] = ldsaddr;                              // lds_addr [63:32]
        g0[2] = (unsigned)ga;                         // global_addr lo
        g0[3] = (unsigned)(ga >> 32) | (2u << 30);    // global_addr hi [120:96] | type=2 [127:126]
        i32x8 g1; for (int t = 0; t < 8; ++t) g1[t] = 0;
        g1[0] = 0x20000;                              // data_size=2 (4B), workgroup_mask=0
        g1[1] = (int)(256u << 16);                    // tensor_dim0 = 256   (bits 79:48)
        g1[2] = (int)(256u << 16);                    // tensor_dim1 = 256   (bits 111:80)
        g1[3] = (int)(256u << 16);                    // tile_dim0  = 256   (bits 127:112)
        g1[4] = 256;                                  // tile_dim1  = 256   (bits 143:128)
        g1[5] = 256;                                  // tensor_dim0_stride = 256 (bits 207:160)
        i32x4 gz; for (int t = 0; t < 4; ++t) gz[t] = 0;
#if __clang_major__ >= 23
        i32x8 z8; for (int t = 0; t < 8; ++t) z8[t] = 0;
        __builtin_amdgcn_tensor_load_to_lds(g0, g1, gz, gz, z8, 0);
#else
        __builtin_amdgcn_tensor_load_to_lds(g0, g1, gz, gz, 0);
#endif
        __builtin_amdgcn_s_wait_tensorcnt(0);
        __syncthreads();
        for (int idx = lane; idx < CST_ELEMS; idx += 32)
            cst[idx] = fminf(cst[idx], MAXC);
    }
#else
    for (int idx = lane; idx < CST_ELEMS; idx += 32)
        cst[idx] = fminf(Db[idx], MAXC);
#endif

    for (int idx = lane; idx < NN + 1; idx += 32) { u[idx] = 0.0f; p[idx] = 0; way[idx] = 0; }
    __syncthreads();

    // Per-lane column state: lane owns columns j = 1 + lane + 32*k, k = 0..7
    float vloc[8];
#pragma unroll
    for (int k = 0; k < 8; ++k) vloc[k] = 0.0f;

    const float INF = 1e30f;

    for (int i = 1; i <= NN; ++i) {
        if (lane == 0) p[0] = i;
        __syncthreads();

        float minv[8];
        bool  used[8];
#pragma unroll
        for (int k = 0; k < 8; ++k) { minv[k] = INF; used[k] = false; }
        bool used0 = false;
        int  j0 = 0;

        for (;;) {
            // used[j0] = true
            if (j0 == 0) {
                used0 = true;
            } else {
                int jm1 = j0 - 1;
                if ((jm1 & 31) == lane) used[jm1 >> 5] = true;
            }
            int   i0  = p[j0];                 // uniform LDS broadcast read
            float ui0 = u[i0];
            const float* row = cst + (size_t)(i0 - 1) * NN;

            // scan free columns: update minv/way, find lane-local argmin
            float best  = INF;
            int   bestj = 1 << 30;
#pragma unroll
            for (int k = 0; k < 8; ++k) {
                if (!used[k]) {
                    int   j   = 1 + lane + (k << 5);
                    float cur = row[j - 1] - ui0 - vloc[k];
                    if (cur < minv[k]) { minv[k] = cur; way[j] = j0; }
                    if (minv[k] < best) { best = minv[k]; bestj = j; } // strict '<' keeps smallest j on ties
                }
            }
            // wave32 argmin reduction (ties -> smallest column index, as np.argmin)
#pragma unroll
            for (int off = 16; off > 0; off >>= 1) {
                float ob = __shfl_xor(best, off, 32);
                int   oj = __shfl_xor(bestj, off, 32);
                if (ob < best || (ob == best && oj < bestj)) { best = ob; bestj = oj; }
            }
            float delta = best;
            int   j1    = bestj;

            // dual updates: u[p[used]] += delta ; v[used] -= delta ; minv[free] -= delta
            if (lane == 0 && used0) u[p[0]] += delta;   // p[0] == i
#pragma unroll
            for (int k = 0; k < 8; ++k) {
                if (used[k]) {
                    int j = 1 + lane + (k << 5);
                    u[p[j]] += delta;                   // distinct rows per used col -> no conflict
                    vloc[k] -= delta;
                } else {
                    minv[k] -= delta;
                }
            }
            __syncthreads();
            j0 = j1;
            if (p[j0] == 0) break;                      // reached a free column
        }

        // augment along 'way' chain (serial, lane 0)
        __syncthreads();
        if (lane == 0) {
            int jj = j0;
            while (jj) { int jn = way[jj]; p[jj] = p[jn]; jj = jn; }
        }
        __syncthreads();
    }

    // col_of_row[p[j]-1] = j-1
#pragma unroll
    for (int k = 0; k < 8; ++k) {
        int j = 1 + lane + (k << 5);
        colofrow[p[j] - 1] = j - 1;
    }
    __syncthreads();

    // gather ORIGINAL (unclamped) matched costs to workspace
#pragma unroll
    for (int k = 0; k < 8; ++k) {
        int r = lane + (k << 5);
        int c = colofrow[r];
        matched_ws[b * NN + r] = Db[(size_t)r * NN + c];
    }
}

// ---------------------------------------------------------------------------
// WMMA mean-reduction of 16*256 = 4096 matched values.
// B = all-ones f16 => C[m][n] accumulates rowsum(A_m); every column of C
// equals the global sum, so sum the 8 accumulator regs and fold lane^16.
// ---------------------------------------------------------------------------
__global__ __launch_bounds__(32)
void wmma_mean_kernel(const float* __restrict__ ws, float* __restrict__ out) {
    const int lane = threadIdx.x;

    v16h ones;
#pragma unroll
    for (int t = 0; t < 16; ++t) ones[t] = (_Float16)1.0f;

    v8f acc;
#pragma unroll
    for (int t = 0; t < 8; ++t) acc[t] = 0.0f;

    // 4096 values / (512 per 16x32 A tile) = 8 WMMA accumulations.
    for (int chunk = 0; chunk < 8; ++chunk) {
        const float* base = ws + chunk * 512 + lane * 16;
        v16h a;
#pragma unroll
        for (int t = 0; t < 16; ++t) a[t] = (_Float16)base[t];
        acc = __builtin_amdgcn_wmma_f32_16x16x32_f16(
            /*neg_a=*/false, a, /*neg_b=*/false, ones,
            /*c_mod=*/(short)0, acc, /*reuse_a=*/false, /*reuse_b=*/false);
    }

    float s = 0.0f;
#pragma unroll
    for (int t = 0; t < 8; ++t) s += acc[t];     // column 'n = lane%16', rows m=0..7 (or 8..15)
    s += __shfl_xor(s, 16, 32);                  // combine m=0..7 with m=8..15 halves

    if (lane == 0) out[0] = s * (1.0f / 4096.0f);  // mean over N then over B == /4096
}

extern "C" void kernel_launch(void* const* d_in, const int* in_sizes, int n_in,
                              void* d_out, int out_size, void* d_ws, size_t ws_size,
                              hipStream_t stream) {
    (void)in_sizes; (void)n_in; (void)out_size; (void)ws_size;
    const float* D  = (const float*)d_in[0];
    float*       out = (float*)d_out;
    float*       ws  = (float*)d_ws;   // 16*256 floats = 16KB of scratch

    hungarian_kernel<<<dim3(BATCH), dim3(32), SMEM_BYTES, stream>>>(D, ws);
    wmma_mean_kernel<<<dim3(1), dim3(32), 0, stream>>>(ws, out);
}
